// FixedPointLinear_32323923870275
// MI455X (gfx1250) — 8-bit matrix operand order repaired and checked by reading, compiled, not run on hardware
//
#include <hip/hip_runtime.h>


typedef __attribute__((ext_vector_type(8))) int v8i;
typedef int vec4i __attribute__((vector_size(16)));

#define QMIN (-32768LL)
#define QMAX (32767LL)

constexpr int M = 4096;
constexpr int N = 4096;
constexpr int K = 4096;

constexpr int BM = 64;
constexpr int BN = 128;
constexpr int BK = 64;
constexpr int LDA = BK + 16;
constexpr int LDB = BK + 16;

#if __has_builtin(__builtin_amdgcn_global_load_async_to_lds_b128)
#define HAVE_ASYNC_COPY 1
#else
#define HAVE_ASYNC_COPY 0
#endif

#if HAVE_ASYNC_COPY
#define ASYNC_COPY16(dst_lds, src_glb)                                      \
  __builtin_amdgcn_global_load_async_to_lds_b128(                           \
      (__attribute__((address_space(1))) vec4i*)(unsigned long long)(src_glb), \
      (__attribute__((address_space(3))) vec4i*)(unsigned long long)(dst_lds), \
      0, 0)
#endif

#if __has_builtin(__builtin_amdgcn_s_wait_asynccnt)
#define WAIT_ASYNC(n) __builtin_amdgcn_s_wait_asynccnt(n)
#else
#define WAIT_ASYNC(n) asm volatile("s_wait_asynccnt %0" ::"i"(n) : "memory")
#endif

__global__ void fxp_quant_kernel(const float4* __restrict__ in,
                                 char4* __restrict__ hi,
                                 uchar4* __restrict__ lo, int n4) {
  int i = blockIdx.x * blockDim.x + threadIdx.x;
  if (i >= n4) return;
  float4 v = in[i];
  int q0 = (int)rintf(fminf(fmaxf(v.x * 256.0f, -32768.0f), 32767.0f));
  int q1 = (int)rintf(fminf(fmaxf(v.y * 256.0f, -32768.0f), 32767.0f));
  int q2 = (int)rintf(fminf(fmaxf(v.z * 256.0f, -32768.0f), 32767.0f));
  int q3 = (int)rintf(fminf(fmaxf(v.w * 256.0f, -32768.0f), 32767.0f));
  char4 h; uchar4 l;
  h.x = (char)(q0 >> 8); l.x = (unsigned char)(q0 & 0xFF);
  h.y = (char)(q1 >> 8); l.y = (unsigned char)(q1 & 0xFF);
  h.z = (char)(q2 >> 8); l.z = (unsigned char)(q2 & 0xFF);
  h.w = (char)(q3 >> 8); l.w = (unsigned char)(q3 & 0xFF);
  hi[i] = h;
  lo[i] = l;
}

__device__ inline v8i load_a_frag(const unsigned char* __restrict__ s,
                                  int rowBase, int lane) {
  const unsigned char* p = s + (rowBase + (lane & 15)) * LDA + ((lane >> 4) << 4);
  int2 d0 = *(const int2*)(p + 0);
  int2 d1 = *(const int2*)(p + 8);
  int2 d2 = *(const int2*)(p + 32);
  int2 d3 = *(const int2*)(p + 40);
  v8i a = {d0.x, d0.y, d1.x, d1.y, d2.x, d2.y, d3.x, d3.y};
  return a;
}

__device__ inline v8i load_b_frag(const unsigned char* __restrict__ s,
                                  int colBase, int lane) {
  const unsigned char* p = s + (colBase + (lane & 15)) * LDB + ((lane >> 4) << 4);
  int4 q0 = *(const int4*)(p + 0);
  int4 q1 = *(const int4*)(p + 32);
  v8i b = {q0.x, q0.y, q0.z, q0.w, q1.x, q1.y, q1.z, q1.w};
  return b;
}

__global__ __launch_bounds__(256) void fxp_gemm_kernel(
    const unsigned char* __restrict__ Ahi, const unsigned char* __restrict__ Alo,
    const unsigned char* __restrict__ Whi, const unsigned char* __restrict__ Wlo,
    const float* __restrict__ bias, float* __restrict__ out) {
  __shared__ __align__(16) unsigned char sAhi[2][BM * LDA];
  __shared__ __align__(16) unsigned char sAlo[2][BM * LDA];
  __shared__ __align__(16) unsigned char sBhi[2][BN * LDB];
  __shared__ __align__(16) unsigned char sBlo[2][BN * LDB];

  const int tid  = threadIdx.x;
  const int lane = tid & 31;
  const int wave = tid >> 5;
  const int wm   = (wave >> 2) * 32;
  const int wn   = (wave & 3) * 32;

  const int m0 = blockIdx.y * BM;
  const int n0 = blockIdx.x * BN;

  const int sr = tid >> 2;
  const int sc = (tid & 3) * 16;

  const size_t aOff  = (size_t)(m0 + sr) * K + sc;
  const size_t b0Off = (size_t)(n0 + sr) * K + sc;
  const size_t b1Off = (size_t)(n0 + sr + 64) * K + sc;
  const int sAoff = sr * LDA + sc;
  const int sB0off = sr * LDB + sc;
  const int sB1off = (sr + 64) * LDB + sc;

  auto stage = [&](int buf, int k0) {
#if HAVE_ASYNC_COPY
    ASYNC_COPY16(&sAhi[buf][sAoff],  Ahi + aOff + k0);
    ASYNC_COPY16(&sAlo[buf][sAoff],  Alo + aOff + k0);
    ASYNC_COPY16(&sBhi[buf][sB0off], Whi + b0Off + k0);
    ASYNC_COPY16(&sBlo[buf][sB0off], Wlo + b0Off + k0);
    ASYNC_COPY16(&sBhi[buf][sB1off], Whi + b1Off + k0);
    ASYNC_COPY16(&sBlo[buf][sB1off], Wlo + b1Off + k0);
#else
    *(int4*)&sAhi[buf][sAoff]  = *(const int4*)(Ahi + aOff + k0);
    *(int4*)&sAlo[buf][sAoff]  = *(const int4*)(Alo + aOff + k0);
    *(int4*)&sBhi[buf][sB0off] = *(const int4*)(Whi + b0Off + k0);
    *(int4*)&sBlo[buf][sB0off] = *(const int4*)(Wlo + b0Off + k0);
    *(int4*)&sBhi[buf][sB1off] = *(const int4*)(Whi + b1Off + k0);
    *(int4*)&sBlo[buf][sB1off] = *(const int4*)(Wlo + b1Off + k0);
#endif
  };

  v8i accHH[2][2], accMID[2][2], accLL[2][2];
  const v8i vzero = {0, 0, 0, 0, 0, 0, 0, 0};
#pragma unroll
  for (int mi = 0; mi < 2; ++mi)
#pragma unroll
    for (int ni = 0; ni < 2; ++ni) {
      accHH[mi][ni] = vzero;
      accMID[mi][ni] = vzero;
      accLL[mi][ni] = vzero;
    }

  stage(0, 0);

  for (int k0 = 0; k0 < K; k0 += BK) {
    const int cur = (k0 >> 6) & 1;

    if (k0 + BK < K) {
      stage(cur ^ 1, k0 + BK);
      WAIT_ASYNC(6);
    } else {
      WAIT_ASYNC(0);
    }
    __syncthreads();

    v8i aLo[2], aHi[2], bLo[2], bHi[2];
#pragma unroll
    for (int mi = 0; mi < 2; ++mi) {
      aLo[mi] = load_a_frag(sAlo[cur], wm + mi * 16, lane);
      aHi[mi] = load_a_frag(sAhi[cur], wm + mi * 16, lane);
    }
#pragma unroll
    for (int ni = 0; ni < 2; ++ni) {
      bLo[ni] = load_b_frag(sBlo[cur], wn + ni * 16, lane);
      bHi[ni] = load_b_frag(sBhi[cur], wn + ni * 16, lane);
    }

#pragma unroll
    for (int mi = 0; mi < 2; ++mi) {
#pragma unroll
      for (int ni = 0; ni < 2; ++ni) {
        accLL[mi][ni] = __builtin_amdgcn_wmma_i32_16x16x64_iu8(
            false, aLo[mi], false, bLo[ni], accLL[mi][ni], false, false);
        accMID[mi][ni] = __builtin_amdgcn_wmma_i32_16x16x64_iu8(
            true, aHi[mi], false, bLo[ni], accMID[mi][ni], false, false);
        accMID[mi][ni] = __builtin_amdgcn_wmma_i32_16x16x64_iu8(
            false, aLo[mi], true, bHi[ni], accMID[mi][ni], false, false);
        accHH[mi][ni] = __builtin_amdgcn_wmma_i32_16x16x64_iu8(
            true, aHi[mi], true, bHi[ni], accHH[mi][ni], false, false);
      }
    }

    __syncthreads();
  }

  const int half = lane >> 4;
  const int ln = lane & 15;
#pragma unroll
  for (int ni = 0; ni < 2; ++ni) {
    const int col = n0 + wn + ni * 16 + ln;
    float bv = fminf(fmaxf(bias[col] * 256.0f, -32768.0f), 32767.0f);
    const long long bq = (long long)rintf(bv);
#pragma unroll
    for (int mi = 0; mi < 2; ++mi) {
#pragma unroll
      for (int r = 0; r < 8; ++r) {
        const int row = m0 + wm + mi * 16 + r + half * 8;
        long long tot = ((long long)accHH[mi][ni][r] << 16) +
                        ((long long)accMID[mi][ni][r] << 8) +
                        (long long)accLL[mi][ni][r];
        long long s = tot >> 8;
        s = s < QMIN ? QMIN : (s > QMAX ? QMAX : s);
        s += bq;
        s = s < QMIN ? QMIN : (s > QMAX ? QMAX : s);
        out[(size_t)row * N + col] = (float)s * (1.0f / 256.0f);
      }
    }
  }
}

extern "C" void kernel_launch(void* const* d_in, const int* in_sizes, int n_in,
                              void* d_out, int out_size, void* d_ws, size_t ws_size,
                              hipStream_t stream) {
  const float* x    = (const float*)d_in[0];
  const float* w    = (const float*)d_in[1];
  const float* bias = (const float*)d_in[2];

  const size_t plane = (size_t)M * K;
  unsigned char* ws  = (unsigned char*)d_ws;
  unsigned char* Ahi = ws;
  unsigned char* Alo = ws + plane;
  unsigned char* Whi = ws + 2 * plane;
  unsigned char* Wlo = ws + 3 * plane;

  {
    int n4 = (int)(plane / 4);
    int blocks = (n4 + 255) / 256;
    fxp_quant_kernel<<<blocks, 256, 0, stream>>>(
        (const float4*)x, (char4*)Ahi, (uchar4*)Alo, n4);
    fxp_quant_kernel<<<blocks, 256, 0, stream>>>(
        (const float4*)w, (char4*)Whi, (uchar4*)Wlo, n4);
  }

  dim3 grid(N / BN, M / BM);
  fxp_gemm_kernel<<<grid, 256, 0, stream>>>(Ahi, Alo, Whi, Wlo, bias,
                                            (float*)d_out);
}
